// GAT_75110388072561
// MI455X (gfx1250) — matrix operand order repaired and checked by reading, not compiled, not run on hardware
//
#include <hip/hip_runtime.h>

typedef __attribute__((ext_vector_type(16))) _Float16 v16h;
typedef __attribute__((ext_vector_type(8)))  _Float16 v8h;
typedef __attribute__((ext_vector_type(8)))  float    v8f;

#define NG 128
#define MT 4   // M-tiles per block in the fused GEMM

__device__ __forceinline__ float lrelu(float v){ return v > 0.f ? v : 0.2f*v; }

__device__ __forceinline__ void atomicMaxF(float* addr, float val){
  if (val >= 0.f) atomicMax((int*)addr, __float_as_int(val));
  else            atomicMin((unsigned int*)addr, (unsigned int)__float_as_int(val));
}

// ---------------- f32 -> f16 conversion ----------------
__global__ void f32_to_f16(const float* __restrict__ x, _Float16* __restrict__ y, long n){
  long i = blockIdx.x*(long)blockDim.x + threadIdx.x;
  if (i < n) y[i] = (_Float16)x[i];
}

// ---------------- fused WMMA GEMM:
//   Fs[N,256] = H[N,64] @ Ws[64,256] + bs
//   Fd[N,256] = H[N,64] @ Wd[64,256] + bd
// One block = MT 16-row M tiles; 16 waves = 16 column strips of 16.
// B fragments (Ws and Wd strips) are loaded once and reused for all MT tiles;
// each A fragment pair feeds 4 WMMAs (2 accumulators x 2 k-steps).
__global__ __launch_bounds__(512)
void gemm_dual(const _Float16* __restrict__ H, const _Float16* __restrict__ Ws,
               const _Float16* __restrict__ Wd, const float* __restrict__ bs,
               const float* __restrict__ bd, float* __restrict__ Fs,
               float* __restrict__ Fd, int N){
  const int lane = threadIdx.x & 31;
  const int wave = threadIdx.x >> 5;      // ntile 0..15
  const int half = lane >> 4;             // 0/1
  const int lr   = lane & 15;
  const int col  = wave*16 + lr;

  // B fragments: column `col` of Ws/Wd, K-split per ISA layout (held in regs)
  v16h bs0, bs1, bd0, bd1;
  #pragma unroll
  for (int i = 0; i < 16; i++){
    const int k0 = (i + 8*half + (i >= 8 ? 8 : 0))      * 256 + col;
    const int k1 = (i + 8*half + (i >= 8 ? 8 : 0) + 32) * 256 + col;
    bs0[i] = Ws[k0];  bs1[i] = Ws[k1];
    bd0[i] = Wd[k0];  bd1[i] = Wd[k1];
  }
  const float biasS = bs[col];
  const float biasD = bd[col];

  const long mbase0 = (long)blockIdx.x * (16*MT);
  #pragma unroll
  for (int mt = 0; mt < MT; mt++){
    const long mbase = mbase0 + mt*16;
    if (mbase < N){                          // wave-uniform guard: EXEC stays all-ones
      // prefetch next tile's feature rows while this tile computes
      if (mbase + 16 < N)
        __builtin_prefetch(H + (mbase + 16 + lr)*64, 0, 1);

      // A fragments: row (mbase+lr), two contiguous 8-half (16B) chunks per half-wave
      const _Float16* hrow = H + (mbase + lr) * 64;
      v8h c00 = *(const v8h*)(hrow      + half*8);
      v8h c01 = *(const v8h*)(hrow + 16 + half*8);
      v8h c10 = *(const v8h*)(hrow + 32 + half*8);
      v8h c11 = *(const v8h*)(hrow + 48 + half*8);
      v16h a0, a1;
      #pragma unroll
      for (int i = 0; i < 8; i++){ a0[i]=c00[i]; a0[i+8]=c01[i]; a1[i]=c10[i]; a1[i+8]=c11[i]; }

      v8f accS = {}, accD = {};
      accS = __builtin_amdgcn_wmma_f32_16x16x32_f16(false, a0, false, bs0, (short)0, accS, false, false);
      accS = __builtin_amdgcn_wmma_f32_16x16x32_f16(false, a1, false, bs1, (short)0, accS, false, false);
      accD = __builtin_amdgcn_wmma_f32_16x16x32_f16(false, a0, false, bd0, (short)0, accD, false, false);
      accD = __builtin_amdgcn_wmma_f32_16x16x32_f16(false, a1, false, bd1, (short)0, accD, false, false);

      float* ps = Fs + (mbase + 8*half)*256 + col;
      float* pd = Fd + (mbase + 8*half)*256 + col;
      #pragma unroll
      for (int r = 0; r < 8; r++){
        ps[(long)r*256] = accS[r] + biasS;
        pd[(long)r*256] = accD[r] + biasD;
      }
    }
  }
}

// ---------------- edge-softmax pipeline ----------------
__global__ void init_md(float* __restrict__ m, float* __restrict__ den, int n){
  int i = blockIdx.x*blockDim.x + threadIdx.x;
  if (i < n){ m[i] = __int_as_float(0xFF800000); den[i] = 0.f; }
}

__global__ void edge_logits(const float* __restrict__ fs, const float* __restrict__ fd,
                            const float* __restrict__ attn, const int* __restrict__ src,
                            const int* __restrict__ dst, float* __restrict__ elog,
                            float* __restrict__ mbuf, long EH){
  long i = blockIdx.x*(long)blockDim.x + threadIdx.x;
  if (i >= EH) return;
  const int  h = (int)(i & 3);
  const long e = i >> 2;
  const int  s = src[e], t = dst[e];
  const float4* ps = (const float4*)(fs + (long)s*256 + h*64);
  const float4* pt = (const float4*)(fd + (long)t*256 + h*64);
  const float4* pa = (const float4*)(attn + h*64);
  float acc = 0.f;
  #pragma unroll
  for (int k = 0; k < 16; k++){
    float4 a = ps[k], b = pt[k], w = pa[k];
    acc += w.x*lrelu(a.x+b.x) + w.y*lrelu(a.y+b.y)
         + w.z*lrelu(a.z+b.z) + w.w*lrelu(a.w+b.w);
  }
  elog[i] = acc;
  atomicMaxF(&mbuf[(long)t*4 + h], acc);
}

__global__ void edge_expsum(float* __restrict__ elog, const float* __restrict__ mbuf,
                            float* __restrict__ den, const int* __restrict__ dst, long EH){
  long i = blockIdx.x*(long)blockDim.x + threadIdx.x;
  if (i >= EH) return;
  const int h = (int)(i & 3);
  const int t = dst[i >> 2];
  float ex = expf(elog[i] - mbuf[(long)t*4 + h]);
  elog[i] = ex;
  atomicAdd(&den[(long)t*4 + h], ex);
}

__global__ void edge_scatter(const float* __restrict__ elog, const float* __restrict__ den,
                             const float* __restrict__ fs, const int* __restrict__ src,
                             const int* __restrict__ dst, float* __restrict__ outp, long EH){
  long i = blockIdx.x*(long)blockDim.x + threadIdx.x;
  if (i >= EH) return;
  const int  h = (int)(i & 3);
  const long e = i >> 2;
  const int  s = src[e], t = dst[e];
  const float alpha = elog[i] / den[(long)t*4 + h];
  const float4* ps = (const float4*)(fs + (long)s*256 + h*64);
  float* po = outp + (long)t*256 + h*64;
  #pragma unroll
  for (int k = 0; k < 16; k++){
    float4 v = ps[k];
    atomicAdd(po + 4*k + 0, alpha*v.x);
    atomicAdd(po + 4*k + 1, alpha*v.y);
    atomicAdd(po + 4*k + 2, alpha*v.z);
    atomicAdd(po + 4*k + 3, alpha*v.w);
  }
}

// ---------------- head-mean + BN stats ----------------
__global__ void node_reduce(const float* __restrict__ outp, float* __restrict__ hn,
                            float* __restrict__ stats, int N){
  __shared__ float s1[256];
  __shared__ float s2[256];
  const int d = threadIdx.x & 63;
  const int r = threadIdx.x >> 6;     // 0..3
  float ps = 0.f, ps2 = 0.f;
  const int base = blockIdx.x * 64;
  for (int n0 = r; n0 < 64; n0 += 4){
    int n = base + n0;
    if (n < N){
      const float* row = outp + (long)n*256;
      float v = 0.25f*(fmaxf(row[d],0.f) + fmaxf(row[64+d],0.f)
                      + fmaxf(row[128+d],0.f) + fmaxf(row[192+d],0.f));
      hn[(long)n*64 + d] = v;
      ps += v; ps2 += v*v;
    }
  }
  s1[threadIdx.x] = ps; s2[threadIdx.x] = ps2;
  __syncthreads();
  if (r == 0){
    float a = s1[d]+s1[64+d]+s1[128+d]+s1[192+d];
    float b = s2[d]+s2[64+d]+s2[128+d]+s2[192+d];
    atomicAdd(&stats[d],    a);
    atomicAdd(&stats[64+d], b);
  }
}

__global__ void bn_finalize(float* __restrict__ stats, float invN){
  int d = threadIdx.x;   // 64
  float mu  = stats[d]*invN;
  float var = stats[64+d]*invN - mu*mu;
  stats[128+d] = mu;
  stats[192+d] = rsqrtf(var + 1e-5f);
}

__global__ void bn_apply_pool(const float* __restrict__ hn, const float* __restrict__ stats,
                              const float* __restrict__ g, const float* __restrict__ b,
                              const int* __restrict__ gid, _Float16* __restrict__ h16,
                              float* __restrict__ pooled, long total){
  long i = blockIdx.x*(long)blockDim.x + threadIdx.x;
  if (i >= total) return;
  const int  d = (int)(i & 63);
  const long n = i >> 6;
  float x = (hn[i] - stats[128+d]) * stats[192+d] * g[d] + b[d];
  h16[i] = (_Float16)x;
  atomicAdd(&pooled[(long)gid[n]*64 + d], x);
}

// ---------------- per-layer pooled MLP: relu(pooled@W + b) -> BN -> cat slice ----------------
__global__ void pool_mlp(const float* __restrict__ pooled, const float* __restrict__ Wl,
                         const float* __restrict__ bl, const float* __restrict__ g,
                         const float* __restrict__ bt, float* __restrict__ catb, int coff){
  __shared__ float ys[NG*64];
  const int d = threadIdx.x;              // 0..63
  for (int r = threadIdx.y; r < NG; r += 16){
    float acc = bl[d];
    for (int k = 0; k < 64; k++) acc += pooled[r*64+k]*Wl[k*64+d];
    ys[r*64+d] = fmaxf(acc, 0.f);
  }
  __syncthreads();
  if (threadIdx.y == 0){
    float s = 0.f, s2 = 0.f;
    for (int r = 0; r < NG; r++){ float v = ys[r*64+d]; s += v; s2 += v*v; }
    float mu = s*(1.f/NG);
    float is = rsqrtf(s2*(1.f/NG) - mu*mu + 1e-5f);
    for (int r = 0; r < NG; r++) catb[r*192 + coff + d] = (ys[r*64+d]-mu)*is*g[d] + bt[d];
  }
}

// ---------------- head: relu(cat@bl_W + b) -> BN -> pooled_hh ----------------
__global__ void head_bl(const float* __restrict__ catb, const float* __restrict__ Wb,
                        const float* __restrict__ bb, const float* __restrict__ g,
                        const float* __restrict__ bt, float* __restrict__ hh){
  __shared__ float ys[NG*64];
  const int d = threadIdx.x;
  for (int r = threadIdx.y; r < NG; r += 16){
    float acc = bb[d];
    for (int k = 0; k < 192; k++) acc += catb[r*192+k]*Wb[k*64+d];
    ys[r*64+d] = fmaxf(acc, 0.f);
  }
  __syncthreads();
  if (threadIdx.y == 0){
    float s = 0.f, s2 = 0.f;
    for (int r = 0; r < NG; r++){ float v = ys[r*64+d]; s += v; s2 += v*v; }
    float mu = s*(1.f/NG);
    float is = rsqrtf(s2*(1.f/NG) - mu*mu + 1e-5f);
    for (int r = 0; r < NG; r++) hh[r*64+d] = (ys[r*64+d]-mu)*is*g[d] + bt[d];
  }
}

// ---------------- head: relu(hh@ll_W + b) -> BN -> log_softmax ----------------
__global__ void head_ll(const float* __restrict__ hh, const float* __restrict__ Wl,
                        const float* __restrict__ bl, const float* __restrict__ g,
                        const float* __restrict__ bt, float* __restrict__ outp){
  __shared__ float zs[NG*10];
  __shared__ float mu_s[10], is_s[10];
  const int r = threadIdx.x;   // 0..127
  for (int c = 0; c < 10; c++){
    float acc = bl[c];
    for (int k = 0; k < 64; k++) acc += hh[r*64+k]*Wl[k*10+c];
    zs[r*10+c] = fmaxf(acc, 0.f);
  }
  __syncthreads();
  if (r < 10){
    float s = 0.f, s2 = 0.f;
    for (int q = 0; q < NG; q++){ float v = zs[q*10+r]; s += v; s2 += v*v; }
    float mu = s*(1.f/NG);
    mu_s[r] = mu;
    is_s[r] = rsqrtf(s2*(1.f/NG) - mu*mu + 1e-5f);
  }
  __syncthreads();
  float y[10]; float mx = -1e30f;
  for (int c = 0; c < 10; c++){
    y[c] = (zs[r*10+c]-mu_s[c])*is_s[c]*g[c] + bt[c];
    mx = fmaxf(mx, y[c]);
  }
  float se = 0.f;
  for (int c = 0; c < 10; c++) se += expf(y[c]-mx);
  float lse = logf(se);
  for (int c = 0; c < 10; c++) outp[r*10+c] = y[c] - mx - lse;
}

extern "C" void kernel_launch(void* const* d_in, const int* in_sizes, int n_in,
                              void* d_out, int out_size, void* d_ws, size_t ws_size,
                              hipStream_t stream) {
  (void)in_sizes; (void)n_in; (void)out_size; (void)ws_size;
  const int N = 50000, E = 800000;
  const long EH = (long)E*4;

  const float* feat = (const float*)d_in[0];
  const float* Wsrc = (const float*)d_in[1];
  const float* bsrc = (const float*)d_in[2];
  const float* Wdst = (const float*)d_in[3];
  const float* bdst = (const float*)d_in[4];
  const float* attn = (const float*)d_in[5];
  const float* bn_g = (const float*)d_in[6];
  const float* bn_b = (const float*)d_in[7];
  const float* lp_W = (const float*)d_in[8];
  const float* lp_b = (const float*)d_in[9];
  const float* lp_g = (const float*)d_in[10];
  const float* lp_bt= (const float*)d_in[11];
  const float* bl_W = (const float*)d_in[12];
  const float* bl_b = (const float*)d_in[13];
  const float* bl_g = (const float*)d_in[14];
  const float* bl_bt= (const float*)d_in[15];
  const float* ll_W = (const float*)d_in[16];
  const float* ll_b = (const float*)d_in[17];
  const float* ll_g = (const float*)d_in[18];
  const float* ll_bt= (const float*)d_in[19];
  const int* src = (const int*)d_in[20];
  const int* dst = (const int*)d_in[21];
  const int* gid = (const int*)d_in[22];

  // workspace layout (~123.4 MB)
  char* w = (char*)d_ws;
  _Float16* h16  = (_Float16*)w;  w += (size_t)N*64*2;       // node feats f16
  _Float16* w16s = (_Float16*)w;  w += (size_t)64*256*2;     // Wsrc[i] f16
  _Float16* w16d = (_Float16*)w;  w += (size_t)64*256*2;     // Wdst[i] f16
  float* fs    = (float*)w;       w += (size_t)N*256*4;      // h@Wsrc
  float* fdout = (float*)w;       w += (size_t)N*256*4;      // h@Wdst, later aggregation out
  float* elog  = (float*)w;       w += (size_t)E*4*4;        // edge logits/exp; later hn [N,64]
  float* mbuf  = (float*)w;       w += (size_t)N*4*4;        // segment max
  float* den   = (float*)w;       w += (size_t)N*4*4;        // segment sum
  float* stats = (float*)w;       w += 256*4;                // BN sums/mu/invstd
  float* pooled= (float*)w;       w += (size_t)NG*64*4;      // graph sum-pool
  float* catb  = (float*)w;       w += (size_t)NG*192*4;     // concat of 3 pooled MLPs

  float* outv = (float*)d_out;    // [128,10] log_softmax
  float* hh   = outv + NG*10;     // [128,64] pooled_hh (second output)

  const int EB = (int)((EH + 255)/256);
  const int GB = (N/16 + MT - 1)/MT;   // fused-GEMM blocks

  f32_to_f16<<<(N*64+255)/256, 256, 0, stream>>>(feat, h16, (long)N*64);

  for (int i = 0; i < 3; i++){
    f32_to_f16<<<64, 256, 0, stream>>>(Wsrc + (size_t)i*64*256, w16s, 64*256);
    f32_to_f16<<<64, 256, 0, stream>>>(Wdst + (size_t)i*64*256, w16d, 64*256);
    gemm_dual<<<GB, 512, 0, stream>>>(h16, w16s, w16d, bsrc + i*256, bdst + i*256,
                                      fs, fdout, N);
    init_md<<<(N*4+255)/256, 256, 0, stream>>>(mbuf, den, N*4);
    edge_logits<<<EB, 256, 0, stream>>>(fs, fdout, attn + (size_t)i*4*64, src, dst, elog, mbuf, EH);
    edge_expsum<<<EB, 256, 0, stream>>>(elog, mbuf, den, dst, EH);
    hipMemsetAsync(fdout, 0, (size_t)N*256*4, stream);       // fd dead -> reuse as agg out
    edge_scatter<<<EB, 256, 0, stream>>>(elog, den, fs, src, dst, fdout, EH);
    hipMemsetAsync(stats, 0, 128*4, stream);
    node_reduce<<<(N+63)/64, 256, 0, stream>>>(fdout, elog /* reuse as hn */, stats, N);
    bn_finalize<<<1, 64, 0, stream>>>(stats, 1.0f/(float)N);
    hipMemsetAsync(pooled, 0, (size_t)NG*64*4, stream);
    bn_apply_pool<<<(int)(((long)N*64+255)/256), 256, 0, stream>>>(
        elog, stats, bn_g + i*64, bn_b + i*64, gid, h16, pooled, (long)N*64);
    pool_mlp<<<1, dim3(64,16), 0, stream>>>(pooled, lp_W + (size_t)i*64*64,
                                            lp_b + i*64, lp_g + i*64, lp_bt + i*64, catb, i*64);
  }
  head_bl<<<1, dim3(64,16), 0, stream>>>(catb, bl_W, bl_b, bl_g, bl_bt, hh);
  head_ll<<<1, 128, 0, stream>>>(hh, ll_W, ll_b, ll_g, ll_bt, outv);
}
